// DeepMonocularModel_88613765251555
// MI455X (gfx1250) — matrix operand order repaired and checked by reading, not compiled, not run on hardware
//
#include <hip/hip_runtime.h>
#include <math.h>

typedef __attribute__((ext_vector_type(16))) _Float16 v16h;
typedef __attribute__((ext_vector_type(8)))  _Float16 h8;   // 16 bytes
typedef __attribute__((ext_vector_type(4)))  _Float16 h4;   // 8 bytes
typedef __attribute__((ext_vector_type(8)))  float    v8f;
typedef __attribute__((ext_vector_type(4)))  int      v4i;  // 16 bytes

#define LDT 40  // LDS row stride in halves: 80 bytes (16B aligned)
#define WTS 32  // pretiled-weight global row stride in halves (dense)

// ---- CDNA5 async global->LDS path (guarded; sync b128 copy fallback) ----
#define USE_ASYNC_LDS 0
#if defined(__has_builtin)
#if __has_builtin(__builtin_amdgcn_global_load_async_to_lds_b128) && \
    __has_builtin(__builtin_amdgcn_s_wait_asynccnt)
#undef USE_ASYNC_LDS
#define USE_ASYNC_LDS 1
#endif
#endif

#if USE_ASYNC_LDS
typedef __attribute__((address_space(1))) v4i* gv4i_p;   // global int4*
typedef __attribute__((address_space(3))) v4i* lv4i_p;   // LDS int4*
#endif

__device__ __forceinline__ void async_fence(){
#if USE_ASYNC_LDS
  __builtin_amdgcn_s_wait_asynccnt(0);
#endif
}

// copy one 128x32h weight tile (512 x 16B chunks, 2 per thread, branchless)
// global rows are dense (WTS halves); LDS rows padded (LDT halves)
__device__ __forceinline__ void copy_b_tile(const _Float16* __restrict__ src,
                                            _Float16* dst, int tid){
  #pragma unroll
  for (int r = 0; r < 2; ++r){
    int u = tid + (r << 8);              // 0..511
    int row = u >> 2, c = (u & 3) << 3;  // 4 chunks of 8 halves per row
#if USE_ASYNC_LDS
    _Float16* s = const_cast<_Float16*>(src + row * WTS + c);
    __builtin_amdgcn_global_load_async_to_lds_b128(
        (gv4i_p)s, (lv4i_p)(dst + row * LDT + c), 0, 0);
#else
    *(h8*)(dst + row * LDT + c) = *(const h8*)(src + row * WTS + c);
#endif
  }
}

__device__ __forceinline__ float gelu_f(float x){
  return 0.5f * x * (1.0f + erff(x * 0.70710678118654752440f));
}

// ---- WMMA fragment loads (16x16x32 f16, wave32) ----
__device__ __forceinline__ v16h frag_a_ld(const _Float16* As, int mbase){
  const int lane = threadIdx.x & 31;
  const _Float16* p = As + (mbase + (lane & 15)) * LDT + ((lane >> 4) << 3);
  h8 lo = *(const h8*)p;
  h8 hi = *(const h8*)(p + 16);
  v16h a;
  #pragma unroll
  for (int j = 0; j < 8; ++j) { a[j] = lo[j]; a[8 + j] = hi[j]; }
  return a;
}
__device__ __forceinline__ v16h frag_b_ld(const _Float16* Bs, int nbase){
  const int lane = threadIdx.x & 31;
  const _Float16* p = Bs + (nbase + (lane & 15)) * LDT + ((lane >> 4) << 3);
  h8 lo = *(const h8*)p;
  h8 hi = *(const h8*)(p + 16);
  v16h b;
  #pragma unroll
  for (int j = 0; j < 8; ++j) { b[j] = lo[j]; b[8 + j] = hi[j]; }
  return b;
}
#define WMMA_F16(A_, B_, C_) \
  __builtin_amdgcn_wmma_f32_16x16x32_f16(false, (A_), false, (B_), (short)0, (C_), false, false)

__device__ __forceinline__ void wmma_tile_step(const _Float16* As, const _Float16* Bs,
                                               int wm, int wn, v8f (&acc)[2][4]){
  v16h a0 = frag_a_ld(As, wm);
  v16h a1 = frag_a_ld(As, wm + 16);
  v16h b0 = frag_b_ld(Bs, wn);
  v16h b1 = frag_b_ld(Bs, wn + 16);
  v16h b2 = frag_b_ld(Bs, wn + 32);
  v16h b3 = frag_b_ld(Bs, wn + 48);
  acc[0][0] = WMMA_F16(a0, b0, acc[0][0]);
  acc[0][1] = WMMA_F16(a0, b1, acc[0][1]);
  acc[0][2] = WMMA_F16(a0, b2, acc[0][2]);
  acc[0][3] = WMMA_F16(a0, b3, acc[0][3]);
  acc[1][0] = WMMA_F16(a1, b0, acc[1][0]);
  acc[1][1] = WMMA_F16(a1, b1, acc[1][1]);
  acc[1][2] = WMMA_F16(a1, b2, acc[1][2]);
  acc[1][3] = WMMA_F16(a1, b3, acc[1][3]);
}

// ============================================================================
// 3x3 conv (pad 1) as implicit GEMM.  Xh: (HW,C) token-major f16 per image.
// Wt: pretiled f16 weights [(kk*Npad+n)*WTS + kin], kk = tap*(Cin/32)+chunk.
// Y: (B,Cout,H,W) f32.  Double-buffered LDS, async B copies.
// ============================================================================
template<int ACT>
__global__ void __launch_bounds__(256)
conv3x3_wmma(const _Float16* __restrict__ Xh, const _Float16* __restrict__ Wt,
             const float* __restrict__ bias, float* __restrict__ Y,
             int Cin, int Cout, int Npad, int H, int Wd)
{
  __shared__ __align__(16) _Float16 As[2][128 * LDT];
  __shared__ __align__(16) _Float16 Bs[2][128 * LDT];
  const int tid  = threadIdx.x;
  const int wave = tid >> 5;
  const int wm   = (wave & 3) << 5;
  const int wn   = (wave >> 2) << 6;
  const int m0   = blockIdx.x << 7;
  const int n0   = blockIdx.y << 7;
  const int b    = blockIdx.z;
  const int logW = (Wd == 32) ? 5 : 6;
  const int HW   = H * Wd;
  const _Float16* Xb = Xh + (size_t)b * HW * Cin;

  // two A-stage units per thread: unit0 = tid, unit1 = tid+256
  const int mu0 = tid >> 2, mu1 = mu0 + 64;
  const int ch  = (tid & 3) << 3;            // half offset within 32-chunk
  const int t0i = m0 + mu0, t1i = m0 + mu1;
  const int y0 = t0i >> logW, x0 = t0i & (Wd - 1);
  const int y1 = t1i >> logW, x1 = t1i & (Wd - 1);

  v8f acc[2][4];
  #pragma unroll
  for (int i = 0; i < 2; ++i)
    #pragma unroll
    for (int j = 0; j < 4; ++j) acc[i][j] = {};

  const int cpt = Cin >> 5;              // chunks per tap
  const int kkTotal = 9 * cpt;
  const h8 zero8 = {};

  // ---- prologue: stage kk=0 (tap 0 => dy=-1,dx=-1, kc=0) into buf 0 ----
  {
    int sy = y0 - 1, sx = x0 - 1;
    bool va = (sy >= 0) & (sy < H) & (sx >= 0) & (sx < Wd);
    int tl = sy * Wd + sx; tl = tl < 0 ? 0 : (tl >= HW ? HW - 1 : tl);
    h8 r0 = *(const h8*)(Xb + tl * Cin + ch);
    *(h8*)(&As[0][mu0 * LDT + ch]) = va ? r0 : zero8;
    sy = y1 - 1; sx = x1 - 1;
    bool vb = (sy >= 0) & (sy < H) & (sx >= 0) & (sx < Wd);
    tl = sy * Wd + sx; tl = tl < 0 ? 0 : (tl >= HW ? HW - 1 : tl);
    h8 r1 = *(const h8*)(Xb + tl * Cin + ch);
    *(h8*)(&As[0][mu1 * LDT + ch]) = vb ? r1 : zero8;
    copy_b_tile(Wt + (size_t)n0 * WTS, Bs[0], tid);
    async_fence();
  }
  __syncthreads();

  int tap = 0, kc = 0;
  for (int kk = 0; kk < kkTotal; ++kk){
    const int cur = kk & 1, nxt = cur ^ 1;
    const bool more = (kk + 1) < kkTotal;
    int ntap = tap, nkc = kc + 32;
    if (nkc == Cin){ nkc = 0; ntap = tap + 1; }

    h8 a0 = zero8, a1 = zero8;
    if (more){
      // issue async B copy + A register loads for tile kk+1 (overlap with WMMA)
      copy_b_tile(Wt + ((size_t)(kk + 1) * Npad + n0) * WTS, Bs[nxt], tid);
      const int dy = ntap / 3 - 1, dx = ntap % 3 - 1;
      int sy = y0 + dy, sx = x0 + dx;
      bool va = (sy >= 0) & (sy < H) & (sx >= 0) & (sx < Wd);
      int tl = sy * Wd + sx; tl = tl < 0 ? 0 : (tl >= HW ? HW - 1 : tl);
      h8 r0 = *(const h8*)(Xb + tl * Cin + nkc + ch);
      a0 = va ? r0 : zero8;
      sy = y1 + dy; sx = x1 + dx;
      bool vb = (sy >= 0) & (sy < H) & (sx >= 0) & (sx < Wd);
      tl = sy * Wd + sx; tl = tl < 0 ? 0 : (tl >= HW ? HW - 1 : tl);
      h8 r1 = *(const h8*)(Xb + tl * Cin + nkc + ch);
      a1 = vb ? r1 : zero8;
    }
    wmma_tile_step(As[cur], Bs[cur], wm, wn, acc);
    if (more){
      *(h8*)(&As[nxt][mu0 * LDT + ch]) = a0;
      *(h8*)(&As[nxt][mu1 * LDT + ch]) = a1;
      async_fence();
    }
    __syncthreads();
    tap = ntap; kc = nkc;
  }

  const int lane = tid & 31;
  float* Yb = Y + (size_t)b * Cout * HW;
  #pragma unroll
  for (int i = 0; i < 2; ++i){
    int tb = m0 + wm + (i << 4) + ((lane >> 4) << 3);   // 8 consecutive tokens
    #pragma unroll
    for (int j = 0; j < 4; ++j){
      int chn = n0 + wn + (j << 4) + (lane & 15);
      if (chn >= Cout) continue;
      float bb = bias[chn];
      float vals[8];
      #pragma unroll
      for (int e = 0; e < 8; ++e){
        float v = acc[i][j][e] + bb;
        if (ACT == 1) v = gelu_f(v);
        vals[e] = v;
      }
      float* p = Yb + chn * HW + tb;      // int offsets (fits 31 bits)
      *(float4*)(p)     = make_float4(vals[0], vals[1], vals[2], vals[3]);
      *(float4*)(p + 4) = make_float4(vals[4], vals[5], vals[6], vals[7]);
    }
  }
}

// ============================================================================
// GEMM with pre-tiled f16 weights: C = act(alpha*A@W + bias + addsrc)
// A: f32 (M,K) row-major (K%4==0). Wt pretiled (Npad cols). Double-buffered.
// ============================================================================
template<int ACT>
__global__ void __launch_bounds__(256)
gemm_pre(const float* __restrict__ A, const _Float16* __restrict__ Wt,
         const float* __restrict__ bias, const float* addsrc, float* C,
         int M, int K, int N, int Npad, float alpha)
{
  __shared__ __align__(16) _Float16 As[2][128 * LDT];
  __shared__ __align__(16) _Float16 Bs[2][128 * LDT];
  const int tid  = threadIdx.x;
  const int wave = tid >> 5;
  const int wm   = (wave & 3) << 5;
  const int wn   = (wave >> 2) << 6;
  const int m0   = blockIdx.x << 7;
  const int n0   = blockIdx.y << 7;

  const int mb = tid >> 3;               // 0..31 (+32*r)
  const int k4 = (tid & 7) << 2;         // 0..28

  v8f acc[2][4];
  #pragma unroll
  for (int i = 0; i < 2; ++i)
    #pragma unroll
    for (int j = 0; j < 4; ++j) acc[i][j] = {};

  const int ksteps = (K + 31) >> 5;

  // ---- prologue: stage ks=0 ----
  {
    #pragma unroll
    for (int r = 0; r < 4; ++r){
      int m = mb + (r << 5);
      int mg = m0 + m, kg = k4;
      int mc = mg < M ? mg : (M - 1);
      int kt = kg > K - 4 ? K - 4 : kg;
      float4 t = *(const float4*)(A + mc * K + kt);
      if (!((mg < M) & (kg < K))) { t.x = 0; t.y = 0; t.z = 0; t.w = 0; }
      h4 hv = { (_Float16)t.x, (_Float16)t.y, (_Float16)t.z, (_Float16)t.w };
      *(h4*)(&As[0][m * LDT + k4]) = hv;
    }
    copy_b_tile(Wt + (size_t)n0 * WTS, Bs[0], tid);
    async_fence();
  }
  __syncthreads();

  for (int ks = 0; ks < ksteps; ++ks){
    const int cur = ks & 1, nxt = cur ^ 1;
    const bool more = (ks + 1) < ksteps;
    float4 ar[4];
    if (more){
      copy_b_tile(Wt + ((size_t)(ks + 1) * Npad + n0) * WTS, Bs[nxt], tid);
      const int c0 = (ks + 1) << 5;
      #pragma unroll
      for (int r = 0; r < 4; ++r){
        int m = mb + (r << 5);
        int mg = m0 + m, kg = c0 + k4;
        int mc = mg < M ? mg : (M - 1);
        int kt = kg > K - 4 ? K - 4 : kg;
        float4 t = *(const float4*)(A + mc * K + kt);
        if (!((mg < M) & (kg < K))) { t.x = 0; t.y = 0; t.z = 0; t.w = 0; }
        ar[r] = t;
      }
    }
    wmma_tile_step(As[cur], Bs[cur], wm, wn, acc);
    if (more){
      #pragma unroll
      for (int r = 0; r < 4; ++r){
        int m = mb + (r << 5);
        h4 hv = { (_Float16)ar[r].x, (_Float16)ar[r].y, (_Float16)ar[r].z, (_Float16)ar[r].w };
        *(h4*)(&As[nxt][m * LDT + k4]) = hv;
      }
      async_fence();
    }
    __syncthreads();
  }

  const int lane = tid & 31;
  #pragma unroll
  for (int i = 0; i < 2; ++i){
    int mt = m0 + wm + (i << 4) + ((lane >> 4) << 3);
    #pragma unroll
    for (int j = 0; j < 4; ++j){
      int nc = n0 + wn + (j << 4) + (lane & 15);
      if (nc >= N) continue;
      float bv = bias ? bias[nc] : 0.0f;
      #pragma unroll
      for (int e = 0; e < 8; ++e){
        int mr = mt + e;
        if (mr >= M) continue;
        int off = mr * N + nc;            // int offsets (fits 31 bits)
        float v = acc[i][j][e] * alpha + bv;
        if (addsrc) v += addsrc[off];
        if (ACT == 1) v = gelu_f(v);
        C[off] = v;
      }
    }
  }
}

// ============================================================================
// Dynamic GEMM (runtime f32 A and W), batched via blockIdx.z. Small problems
// (attention logits/ctx) => single-buffered, branchless clamped staging.
// ============================================================================
template<int ACT>
__global__ void __launch_bounds__(256)
gemm_dyn(const float* __restrict__ A, const float* __restrict__ W,
         const float* __restrict__ bias, const float* addsrc, float* C,
         int M, int K, int N,
         long long sA, long long sW, long long sC, float alpha)
{
  __shared__ __align__(16) _Float16 As[128 * LDT];
  __shared__ __align__(16) _Float16 Bs[128 * LDT];
  const int tid  = threadIdx.x;
  const int wave = tid >> 5;
  const int wm   = (wave & 3) << 5;
  const int wn   = (wave >> 2) << 6;
  const int m0   = blockIdx.x << 7;
  const int n0   = blockIdx.y << 7;
  const long long bz = blockIdx.z;
  const float* Ab = A + bz * sA;
  const float* Wb = W + bz * sW;
  float* Cb = C + bz * sC;
  const float* asrc = addsrc ? addsrc + bz * sC : nullptr;

  const int mb = tid >> 3;
  const int k4 = (tid & 7) << 2;
  const int nB = tid & 127;              // B staging: one column per thread
  const int kB = (tid >> 7) << 4;        // 16 k per thread

  v8f acc[2][4];
  #pragma unroll
  for (int i = 0; i < 2; ++i)
    #pragma unroll
    for (int j = 0; j < 4; ++j) acc[i][j] = {};

  const int ksteps = (K + 31) >> 5;
  for (int ks = 0; ks < ksteps; ++ks){
    const int c0 = ks << 5;
    #pragma unroll
    for (int r = 0; r < 4; ++r){         // A tile: float4 per unit, clamped
      int m = mb + (r << 5);
      int mg = m0 + m, kg = c0 + k4;
      int mc = mg < M ? mg : (M - 1);
      int kt = kg > K - 4 ? K - 4 : kg;
      float4 t = *(const float4*)(Ab + mc * K + kt);
      if (!((mg < M) & (kg < K))) { t.x = 0; t.y = 0; t.z = 0; t.w = 0; }
      h4 hv = { (_Float16)t.x, (_Float16)t.y, (_Float16)t.z, (_Float16)t.w };
      *(h4*)(&As[m * LDT + k4]) = hv;
    }
    {                                    // B tile: 16 k-values for one column
      _Float16 hb[16];
      int ng = n0 + nB;
      int ncl = ng < N ? ng : (N - 1);
      #pragma unroll
      for (int j = 0; j < 16; ++j){
        int kg = c0 + kB + j;
        int kc = kg < K ? kg : (K - 1);
        float raw = Wb[kc * N + ncl];
        hb[j] = (_Float16)(((kg < K) & (ng < N)) ? raw : 0.0f);
      }
      h8 lo, hi;
      #pragma unroll
      for (int j = 0; j < 8; ++j){ lo[j] = hb[j]; hi[j] = hb[8 + j]; }
      *(h8*)(&Bs[nB * LDT + kB])     = lo;
      *(h8*)(&Bs[nB * LDT + kB + 8]) = hi;
    }
    __syncthreads();
    wmma_tile_step(As, Bs, wm, wn, acc);
    __syncthreads();
  }

  const int lane = tid & 31;
  #pragma unroll
  for (int i = 0; i < 2; ++i){
    int mt = m0 + wm + (i << 4) + ((lane >> 4) << 3);
    #pragma unroll
    for (int j = 0; j < 4; ++j){
      int nc = n0 + wn + (j << 4) + (lane & 15);
      if (nc >= N) continue;
      float bv = bias ? bias[nc] : 0.0f;
      #pragma unroll
      for (int e = 0; e < 8; ++e){
        int mr = mt + e;
        if (mr >= M) continue;
        int off = mr * N + nc;
        float v = acc[i][j][e] * alpha + bv;
        if (asrc) v += asrc[off];
        if (ACT == 1) v = gelu_f(v);
        Cb[off] = v;
      }
    }
  }
}

// ---- pretile GEMM weights: f32 (K,N) -> dense f16 tiles [(kc*Npad+n)*WTS+kin] ----
__global__ void __launch_bounds__(256)
pretile_w_kernel(const float* __restrict__ W, _Float16* __restrict__ Wt,
                 int K, int N, int Npad)
{
  long long idx = (long long)blockIdx.x * 256 + threadIdx.x;
  long long total = (long long)((K + 31) >> 5) * Npad * WTS;
  if (idx >= total) return;
  int j = (int)(idx & 31);
  long long r = idx >> 5;
  int n  = (int)(r % Npad);
  int kc = (int)(r / Npad);
  int k  = (kc << 5) + j;
  float v = 0.0f;
  if (k < K && n < N) v = W[(long long)k * N + n];
  Wt[idx] = (_Float16)v;
}

// ---- pretile conv weights: (O,I,3,3) -> dense f16 tiles, k = tap*Cin + i ----
__global__ void __launch_bounds__(256)
pretile_conv_w_kernel(const float* __restrict__ w, _Float16* __restrict__ Wt,
                      int Cin, int Cout, int Npad)
{
  long long idx = (long long)blockIdx.x * 256 + threadIdx.x;
  long long total = (long long)((9 * Cin) >> 5) * Npad * WTS;
  if (idx >= total) return;
  int j = (int)(idx & 31);
  long long r = idx >> 5;
  int n  = (int)(r % Npad);
  int kc = (int)(r / Npad);
  float v = 0.0f;
  if (n < Cout){
    int k = (kc << 5) + j;
    int tap = k / Cin, i = k % Cin;
    v = w[((long long)n * Cin + i) * 9 + tap];
  }
  Wt[idx] = (_Float16)v;
}

// ---- prestage activations: (B,C,Hs,Ws) f32 chan-major -> (B,HW,C) f16 token-major
//      UP=1: nearest-neighbor 2x upsample fused (H=2*Hs, W=2*Ws) ----
template<int UP>
__global__ void __launch_bounds__(256)
prestage_x_kernel(const float* __restrict__ X, _Float16* __restrict__ Xh,
                  int C, int Hs, int Ws)
{
  __shared__ float tile[32][33];
  const int Wd = UP ? (Ws << 1) : Ws;
  const int H  = UP ? (Hs << 1) : Hs;
  const int HW = H * Wd;
  const int HsWs = Hs * Ws;
  const int lw = (Wd == 64) ? 6 : 5;
  int b = blockIdx.x, t0 = blockIdx.y << 5, c0 = blockIdx.z << 5;
  int lj = threadIdx.x & 31, r0 = threadIdx.x >> 5;
  #pragma unroll
  for (int r = 0; r < 4; ++r){
    int ci = r0 + (r << 3);
    int t = t0 + lj;
    int src = t;
    if (UP){ int y = t >> lw, x = t & (Wd - 1); src = (y >> 1) * Ws + (x >> 1); }
    tile[ci][lj] = X[((size_t)b * C + c0 + ci) * HsWs + src];
  }
  __syncthreads();
  #pragma unroll
  for (int r = 0; r < 4; ++r){
    int ti = r0 + (r << 3);
    Xh[((size_t)b * HW + t0 + ti) * C + c0 + lj] = (_Float16)tile[lj][ti];
  }
}

// ---- tokens = transpose(feats) + pos_enc (f32, for attention) ----
__global__ void __launch_bounds__(256)
build_tokens_kernel(const float* __restrict__ feats, const float* __restrict__ pos,
                    float* __restrict__ tokens)
{
  __shared__ float tile[32][33];
  int b = blockIdx.x, t0 = blockIdx.y << 5, c0 = blockIdx.z << 5;
  int lj = threadIdx.x & 31, r0 = threadIdx.x >> 5;
  #pragma unroll
  for (int r = 0; r < 4; ++r){
    int ci = r0 + (r << 3);
    tile[ci][lj] = feats[((size_t)b * 768 + c0 + ci) * 1024 + t0 + lj];
  }
  __syncthreads();
  #pragma unroll
  for (int r = 0; r < 4; ++r){
    int ti = r0 + (r << 3);
    tokens[((size_t)b * 1024 + t0 + ti) * 768 + c0 + lj] =
        tile[lj][ti] + pos[(size_t)(t0 + ti) * 768 + c0 + lj];
  }
}

// ---- depth head: fused up2 + 1x1 conv + softplus ----
__global__ void __launch_bounds__(256)
depth_head_kernel(const float* __restrict__ d2, const float* __restrict__ w,
                  const float* __restrict__ bias, float* __restrict__ out)
{
  int i = blockIdx.x * 256 + threadIdx.x;
  if (i >= 32 * 128 * 128) return;
  int b = i >> 14, rem = i & 16383, yy = rem >> 7, xx = rem & 127;
  const float* src = d2 + ((size_t)b * 32 * 64 + (yy >> 1)) * 64 + (xx >> 1);
  float a = bias[0];
  #pragma unroll
  for (int c = 0; c < 32; ++c) a += src[(size_t)c * 4096] * w[c];
  out[i] = (a > 20.0f) ? a : log1pf(expf(a));
}

// ---- q0 = [onehot(intent-1,3), past.flat(96)] @ qi_w + qi_b ----
__global__ void __launch_bounds__(256)
init_query_kernel(const int* __restrict__ intent, const float* __restrict__ past,
                  const float* __restrict__ qi_w, const float* __restrict__ qi_b,
                  float* __restrict__ q)
{
  __shared__ float inp[99];
  int b = blockIdx.x, tid = threadIdx.x;
  if (tid < 99)
    inp[tid] = (tid < 3) ? ((intent[b] - 1 == tid) ? 1.0f : 0.0f) : past[b * 96 + tid - 3];
  __syncthreads();
  for (int n = tid; n < 768; n += 256){
    float a = qi_b[n];
    for (int k = 0; k < 99; ++k) a += inp[k] * qi_w[k * 768 + n];
    q[(size_t)b * 768 + n] = a;
  }
}

// ---- LayerNorm over 32 rows of 768 ----
__global__ void __launch_bounds__(256)
ln_rows_kernel(const float* __restrict__ x, const float* __restrict__ g,
               const float* __restrict__ bta, float* __restrict__ y)
{
  __shared__ float red[256];
  int b = blockIdx.x, tid = threadIdx.x;
  const float* xr = x + (size_t)b * 768;
  float v0 = xr[tid], v1 = xr[tid + 256], v2 = xr[tid + 512];
  red[tid] = v0 + v1 + v2; __syncthreads();
  for (int s = 128; s > 0; s >>= 1){ if (tid < s) red[tid] += red[tid + s]; __syncthreads(); }
  float mean = red[0] * (1.0f / 768.0f); __syncthreads();
  float d0 = v0 - mean, d1 = v1 - mean, d2 = v2 - mean;
  red[tid] = d0 * d0 + d1 * d1 + d2 * d2; __syncthreads();
  for (int s = 128; s > 0; s >>= 1){ if (tid < s) red[tid] += red[tid + s]; __syncthreads(); }
  float inv = rsqrtf(red[0] * (1.0f / 768.0f) + 1e-5f);
  float* yr = y + (size_t)b * 768;
  yr[tid]       = d0 * inv * g[tid]       + bta[tid];
  yr[tid + 256] = d1 * inv * g[tid + 256] + bta[tid + 256];
  yr[tid + 512] = d2 * inv * g[tid + 512] + bta[tid + 512];
}

// ---- PT[b,c,h] = sum_d wk[c, h*96+d] * Q[b, h*96+d] ----
__global__ void __launch_bounds__(256)
attn_pk_kernel(const float* __restrict__ wk, const float* __restrict__ Qp,
               float* __restrict__ PT)
{
  __shared__ float qs[768];
  int b = blockIdx.x, tid = threadIdx.x;
  for (int i = tid; i < 768; i += 256) qs[i] = Qp[(size_t)b * 768 + i];
  __syncthreads();
  for (int c = tid; c < 768; c += 256){
    const float* wr = wk + (size_t)c * 768;
    float* p = PT + ((size_t)b * 768 + c) * 8;
    #pragma unroll
    for (int h = 0; h < 8; ++h){
      float a = 0.0f;
      #pragma unroll 8
      for (int d = 0; d < 96; ++d) a += wr[h * 96 + d] * qs[h * 96 + d];
      p[h] = a;
    }
  }
}

// ---- softmax over tokens; in (B,1024,8) -> out (B,8,1024) ----
__global__ void __launch_bounds__(256)
softmax_t_kernel(const float* __restrict__ logits, float* __restrict__ att)
{
  __shared__ float red[256];
  int b = blockIdx.x >> 3, h = blockIdx.x & 7, tid = threadIdx.x;
  const float* in = logits + (size_t)b * 8192 + h;
  float* out = att + ((size_t)b * 8 + h) * 1024;
  float v[4];
  float mx = -3.0e38f;
  #pragma unroll
  for (int r = 0; r < 4; ++r){ v[r] = in[(tid + r * 256) * 8]; mx = fmaxf(mx, v[r]); }
  red[tid] = mx; __syncthreads();
  for (int s = 128; s > 0; s >>= 1){ if (tid < s) red[tid] = fmaxf(red[tid], red[tid + s]); __syncthreads(); }
  mx = red[0]; __syncthreads();
  float sum = 0.0f;
  #pragma unroll
  for (int r = 0; r < 4; ++r){ v[r] = expf(v[r] - mx); sum += v[r]; }
  red[tid] = sum; __syncthreads();
  for (int s = 128; s > 0; s >>= 1){ if (tid < s) red[tid] += red[tid + s]; __syncthreads(); }
  float inv = 1.0f / red[0];
  #pragma unroll
  for (int r = 0; r < 4; ++r) out[tid + r * 256] = v[r] * inv;
}

// ---- o[b, h*96+d] = ctx[b,h,:] @ wv[:, h*96+d] + bv ----
__global__ void __launch_bounds__(128)
attn_out_kernel(const float* __restrict__ ctx, const float* __restrict__ wv,
                const float* __restrict__ bv, float* __restrict__ o)
{
  __shared__ float cs[768];
  int b = blockIdx.x, h = blockIdx.y, tid = threadIdx.x;
  for (int i = tid; i < 768; i += 128) cs[i] = ctx[((size_t)b * 8 + h) * 768 + i];
  __syncthreads();
  if (tid < 96){
    float a = bv[h * 96 + tid];
    for (int c = 0; c < 768; ++c) a += cs[c] * wv[(size_t)c * 768 + h * 96 + tid];
    o[(size_t)b * 768 + h * 96 + tid] = a;
  }
}

__global__ void __launch_bounds__(256)
controls_kernel(const float* __restrict__ cp, float* __restrict__ c1, float* __restrict__ c2)
{
  int i = blockIdx.x * 256 + threadIdx.x;
  if (i >= 96000) return;
  float v = tanhf(cp[i]) * ((i & 1) ? 1.0f : 8.0f);
  c1[i] = v; c2[i] = v;
}

__global__ void __launch_bounds__(256)
rollout_kernel(const float* __restrict__ past, const float* __restrict__ ctrl,
               float* __restrict__ t1o, float* __restrict__ t2o)
{
  int i = blockIdx.x * 256 + threadIdx.x;
  if (i >= 1600) return;
  int b = i / 50, k = i % 50;
  const float* p = past + b * 96 + 90;
  float x = p[0], y = p[1], vx = p[2], vy = p[3];
  float s  = sqrtf(vx * vx + vy * vy + 1e-6f);
  float hd = atan2f(vy, vx);
  for (int t = 0; t < 30; ++t){
    size_t ci = (size_t)b * 3000 + (k * 30 + t) * 2;
    float a = ctrl[ci], w = ctrl[ci + 1];
    x += s * cosf(hd) * 0.25f;
    y += s * sinf(hd) * 0.25f;
    t1o[ci] = x; t1o[ci + 1] = y;
    t2o[ci] = x; t2o[ci + 1] = y;
    hd += w * 0.25f;
    s = fmaxf(s + a * 0.25f, 0.0f);
  }
}

__global__ void __launch_bounds__(256)
bcast_q_kernel(const float* __restrict__ qv, float* __restrict__ out)
{
  long long i = (long long)blockIdx.x * 256 + threadIdx.x;
  if (i >= 1228800LL) return;
  long long row = i / 768; int b = (int)(row / 50);
  out[i] = qv[(size_t)b * 768 + (i % 768)];
}

__global__ void __launch_bounds__(256)
score_kernel(const float* __restrict__ x, const float* __restrict__ w,
             const float* __restrict__ bias, float* __restrict__ out)
{
  __shared__ float red[256];
  int row = blockIdx.x, tid = threadIdx.x;
  const float* xr = x + (size_t)row * 768;
  red[tid] = xr[tid] * w[tid] + xr[tid + 256] * w[tid + 256] + xr[tid + 512] * w[tid + 512];
  __syncthreads();
  for (int s = 128; s > 0; s >>= 1){ if (tid < s) red[tid] += red[tid + s]; __syncthreads(); }
  if (tid == 0) out[row] = red[0] + bias[0];
}

// ============================================================================
extern "C" void kernel_launch(void* const* d_in, const int* in_sizes, int n_in,
                              void* d_out, int out_size, void* d_ws, size_t ws_size,
                              hipStream_t stream)
{
  (void)in_sizes; (void)n_in; (void)out_size; (void)ws_size;
  const float* feats_vit = (const float*)d_in[0];
  const float* past      = (const float*)d_in[1];
  const int*   intent    = (const int*)  d_in[2];
  const float* qi_w = (const float*)d_in[3];   const float* qi_b = (const float*)d_in[4];
  const float* va1_w = (const float*)d_in[5];  const float* va1_b = (const float*)d_in[6];
  const float* va2_w = (const float*)d_in[7];  const float* va2_b = (const float*)d_in[8];
  const float* pos_enc = (const float*)d_in[9];
  const float* ln1g = (const float*)d_in[10];  const float* ln1b = (const float*)d_in[11];
  const float* wq = (const float*)d_in[12];    const float* bq = (const float*)d_in[13];
  const float* wk = (const float*)d_in[14];    // bk unused: softmax shift-invariant
  const float* wv = (const float*)d_in[16];    const float* bv = (const float*)d_in[17];
  const float* wo = (const float*)d_in[18];    const float* bo = (const float*)d_in[19];
  const float* ln2g = (const float*)d_in[20];  const float* ln2b = (const float*)d_in[21];
  const float* m1w = (const float*)d_in[22];   const float* m1b = (const float*)d_in[23];
  const float* m2w = (const float*)d_in[24];   const float* m2b = (const float*)d_in[25];
  const float* dg1_w = (const float*)d_in[26]; const float* dg1_b = (const float*)d_in[27];
  const float* dg2_w = (const float*)d_in[28]; const float* dg2_b = (const float*)d_in[29];
  const float* dg3_w = (const float*)d_in[30]; const float* dg3_b = (const float*)d_in[31];
  const float* td1_w = (const float*)d_in[32]; const float* td1_b = (const float*)d_in[33];
  const float* td2_w = (const float*)d_in[34]; const float* td2_b = (const float*)d_in[35];
  const float* td3_w = (const float*)d_in[36]; const float* td3_b = (const float*)d_in[37];
  const float* tf1_w = (const float*)d_in[38]; const float* tf1_b = (const float*)d_in[39];
  const float* tf2_w = (const float*)d_in[40]; const float* tf2_b = (const float*)d_in[41];
  const float* sd1_w = (const float*)d_in[42]; const float* sd1_b = (const float*)d_in[43];
  const float* sd2_w = (const float*)d_in[44]; const float* sd2_b = (const float*)d_in[45];
  const float* sd3_w = (const float*)d_in[46]; const float* sd3_b = (const float*)d_in[47];

  float* out = (float*)d_out;
  float* out_traj_r = out;
  float* out_traj   = out + 96000;
  float* out_qsc    = out + 192000;
  float* out_scores = out + 1420800;
  float* out_depth  = out + 1422400;
  float* out_ctrl_r = out + 1946688;
  float* out_ctrl   = out + 2042688;

  float* ws   = (float*)d_ws;
  float* bufA = ws;                  // 25,165,824 (h1, later tokens)
  float* bufB = bufA + 25165824;     // 25,165,824 (feats)
  float* d1   = bufB + 25165824;     // 2,097,152
  float* d2   = d1   + 2097152;      // 4,194,304
  float* q    = d2   + 4194304;      // 24,576
  float* qn   = q    + 24576;
  float* Qp   = qn   + 24576;
  float* PT   = Qp   + 24576;        // 196,608  (B,768,8)
  float* lg   = PT   + 196608;       // 262,144  (B,1024,8)
  float* att  = lg   + 262144;       // 262,144  (B,8,1024)
  float* ctx  = att  + 262144;       // 196,608  (B,8,768)
  float* ov   = ctx  + 196608;       // 24,576
  float* hm   = ov   + 24576;        // 98,304
  float* th1  = hm   + 98304;        // 24,576
  float* th2  = th1  + 24576;
  float* cp   = th2  + 24576;        // 96,000
  float* tfA  = cp   + 96000;        // 1,228,800
  float* tfB  = tfA  + 1228800;
  float* t1   = tfB  + 1228800;
  float* sc   = t1   + 1228800;
  float* sc2  = sc   + 1228800;
  _Float16* XhA = (_Float16*)(sc2 + 1228800);            // 25,165,824 halves
  _Float16* WtA = (_Float16*)(sc2 + 1228800 + 12582912); // 5,308,416 halves (dense)

  // ---------------- Vision backbone ----------------
  prestage_x_kernel<0><<<dim3(32, 32, 24), 256, 0, stream>>>(feats_vit, XhA, 768, 32, 32);
  pretile_conv_w_kernel<<<20736, 256, 0, stream>>>(va1_w, WtA, 768, 768, 768);
  conv3x3_wmma<1><<<dim3(8, 6, 32), 256, 0, stream>>>(XhA, WtA, va1_b, bufA, 768, 768, 768, 32, 32);
  prestage_x_kernel<0><<<dim3(32, 32, 24), 256, 0, stream>>>(bufA, XhA, 768, 32, 32);
  pretile_conv_w_kernel<<<20736, 256, 0, stream>>>(va2_w, WtA, 768, 768, 768);
  conv3x3_wmma<0><<<dim3(8, 6, 32), 256, 0, stream>>>(XhA, WtA, va2_b, bufB, 768, 768, 768, 32, 32);

  // ---------------- Depth head ----------------
  prestage_x_kernel<0><<<dim3(32, 32, 24), 256, 0, stream>>>(bufB, XhA, 768, 32, 32);
  pretile_conv_w_kernel<<<3456, 256, 0, stream>>>(dg1_w, WtA, 768, 64, 128);
  conv3x3_wmma<1><<<dim3(8, 1, 32), 256, 0, stream>>>(XhA, WtA, dg1_b, d1, 768, 64, 128, 32, 32);
  prestage_x_kernel<1><<<dim3(32, 128, 2), 256, 0, stream>>>(d1, XhA, 64, 32, 32);
  pretile_conv_w_kernel<<<288, 256, 0, stream>>>(dg2_w, WtA, 64, 32, 128);
  conv3x3_wmma<1><<<dim3(32, 1, 32), 256, 0, stream>>>(XhA, WtA, dg2_b, d2, 64, 32, 128, 64, 64);
  depth_head_kernel<<<2048, 256, 0, stream>>>(d2, dg3_w, dg3_b, out_depth);

  // ---------------- Tokens + query ----------------
  build_tokens_kernel<<<dim3(32, 32, 24), 256, 0, stream>>>(bufB, pos_enc, bufA);
  init_query_kernel<<<32, 256, 0, stream>>>(intent, past, qi_w, qi_b, q);

  // ---------------- 4 cross-attention blocks ----------------
  const float scale = 0.10206207261596575f;   // 1/sqrt(96)
  for (int i = 0; i < 4; ++i){
    long long w2 = (long long)i * 768 * 768;
    int bo1 = i * 768;
    ln_rows_kernel<<<32, 256, 0, stream>>>(q, ln1g + bo1, ln1b + bo1, qn);
    pretile_w_kernel<<<2304, 256, 0, stream>>>(wq + w2, WtA, 768, 768, 768);
    gemm_pre<0><<<dim3(1, 6, 1), 256, 0, stream>>>(qn, WtA, bq + bo1, nullptr, Qp,
                                                   32, 768, 768, 768, 1.0f);
    attn_pk_kernel<<<32, 256, 0, stream>>>(wk + w2, Qp, PT);
    gemm_dyn<0><<<dim3(8, 1, 32), 256, 0, stream>>>(bufA, PT, nullptr, nullptr, lg,
                                                    1024, 768, 8, 786432LL, 6144LL, 8192LL, scale);
    softmax_t_kernel<<<256, 256, 0, stream>>>(lg, att);
    gemm_dyn<0><<<dim3(1, 6, 32), 256, 0, stream>>>(att, bufA, nullptr, nullptr, ctx,
                                                    8, 1024, 768, 8192LL, 786432LL, 6144LL, 1.0f);
    attn_out_kernel<<<dim3(32, 8), 128, 0, stream>>>(ctx, wv + w2, bv + bo1, ov);
    pretile_w_kernel<<<2304, 256, 0, stream>>>(wo + w2, WtA, 768, 768, 768);
    gemm_pre<0><<<dim3(1, 6, 1), 256, 0, stream>>>(ov, WtA, bo + bo1, q, q,
                                                   32, 768, 768, 768, 1.0f);
    ln_rows_kernel<<<32, 256, 0, stream>>>(q, ln2g + bo1, ln2b + bo1, qn);
    pretile_w_kernel<<<9216, 256, 0, stream>>>(m1w + (long long)i * 768 * 3072, WtA, 768, 3072, 3072);
    gemm_pre<1><<<dim3(1, 24, 1), 256, 0, stream>>>(qn, WtA, m1b + i * 3072, nullptr, hm,
                                                    32, 768, 3072, 3072, 1.0f);
    pretile_w_kernel<<<9216, 256, 0, stream>>>(m2w + (long long)i * 3072 * 768, WtA, 3072, 768, 768);
    gemm_pre<0><<<dim3(1, 6, 1), 256, 0, stream>>>(hm, WtA, m2b + bo1, q, q,
                                                   32, 3072, 768, 768, 1.0f);
  }

  // ---------------- Trajectory decoder + rollout ----------------
  pretile_w_kernel<<<2304, 256, 0, stream>>>(td1_w, WtA, 768, 768, 768);
  gemm_pre<1><<<dim3(1, 6, 1), 256, 0, stream>>>(q, WtA, td1_b, nullptr, th1, 32, 768, 768, 768, 1.0f);
  pretile_w_kernel<<<2304, 256, 0, stream>>>(td2_w, WtA, 768, 768, 768);
  gemm_pre<1><<<dim3(1, 6, 1), 256, 0, stream>>>(th1, WtA, td2_b, nullptr, th2, 32, 768, 768, 768, 1.0f);
  pretile_w_kernel<<<9216, 256, 0, stream>>>(td3_w, WtA, 768, 3000, 3072);
  gemm_pre<0><<<dim3(1, 24, 1), 256, 0, stream>>>(th2, WtA, td3_b, nullptr, cp, 32, 768, 3000, 3072, 1.0f);
  controls_kernel<<<375, 256, 0, stream>>>(cp, out_ctrl_r, out_ctrl);
  rollout_kernel<<<7, 256, 0, stream>>>(past, out_ctrl, out_traj_r, out_traj);
  bcast_q_kernel<<<4800, 256, 0, stream>>>(q, out_qsc);

  // ---------------- Scorer ----------------
  pretile_w_kernel<<<192, 256, 0, stream>>>(tf1_w, WtA, 60, 768, 768);
  gemm_pre<1><<<dim3(13, 6, 1), 256, 0, stream>>>(out_traj, WtA, tf1_b, nullptr, tfA, 1600, 60, 768, 768, 1.0f);
  pretile_w_kernel<<<2304, 256, 0, stream>>>(tf2_w, WtA, 768, 768, 768);
  gemm_pre<1><<<dim3(13, 6, 1), 256, 0, stream>>>(tfA, WtA, tf2_b, nullptr, tfB, 1600, 768, 768, 768, 1.0f);
  pretile_w_kernel<<<2304, 256, 0, stream>>>(sd1_w, WtA, 768, 768, 768);
  gemm_pre<0><<<dim3(13, 6, 1), 256, 0, stream>>>(out_qsc, WtA, sd1_b, nullptr, t1, 1600, 768, 768, 768, 1.0f);
  pretile_w_kernel<<<2304, 256, 0, stream>>>(sd1_w + 589824, WtA, 768, 768, 768);
  gemm_pre<1><<<dim3(13, 6, 1), 256, 0, stream>>>(tfB, WtA, nullptr, t1, sc, 1600, 768, 768, 768, 1.0f);
  pretile_w_kernel<<<2304, 256, 0, stream>>>(sd2_w, WtA, 768, 768, 768);
  gemm_pre<1><<<dim3(13, 6, 1), 256, 0, stream>>>(sc, WtA, sd2_b, nullptr, sc2, 1600, 768, 768, 768, 1.0f);
  score_kernel<<<1600, 256, 0, stream>>>(sc2, sd3_w, sd3_b, out_scores);
}
